// GaussSynthesis_71347996721427
// MI455X (gfx1250) — matrix operand order repaired and checked by reading, not compiled, not run on hardware
//
#include <hip/hip_runtime.h>
#include <hip/hip_bf16.h>

// GaussSynthesis on MI455X (gfx1250, wave32, WMMA + async global->LDS DMA).
//
// out[m,t] = (2/V) * sum_{k=1..256} Yr[m,k] cos(2pi k t/V) - Yi[m,k] sin(2pi k t/V)
// => out[4096, 50257] = Ys[4096, 512] @ Basis[512, 50257]   (~210 TFLOP, dominant)
// v_wmma_f32_16x16x32_f16 with hi/lo f16 split (hh + hl + lh) for ~fp32 accuracy.
// Basis (hi+lo f16, ~98MB) stays L2-resident; 823MB output write is the floor.
// Staging uses GLOBAL_LOAD_ASYNC_TO_LDS_B128 (ASYNCcnt) -> zero staging VGPRs,
// which removes the scratch spills seen with the reg-relay double buffer.

typedef __attribute__((ext_vector_type(16))) _Float16 v16h;
typedef __attribute__((ext_vector_type(8)))  float    v8f;
typedef int v4i_vs __attribute__((vector_size(4 * sizeof(int))));

#define NFREQ   256
#define VOCAB   50257
#define NCH     1024
#define MTOT    4096            // B*S
#define KTOT    512             // 2*NFREQ
#define KSTEPS  16              // KTOT/32
#define NT16    3144            // ceil-pad columns: 3144*16 = 50304 = 393*128
#define NT128   393
#define MT16    256             // MTOT/16

// ---- CDNA5 async global->LDS copy (16B per lane), ASYNCcnt-tracked ---------
// Builtin signature (from clang diagnostic): param0 = AS1 pointer to
// vector_size(16) int (global source), param1 = AS3 pointer (LDS dest),
// then imm offset + imm cpol.
__device__ __forceinline__ void async_cp16(const void* g, void* l) {
    auto gp = (__attribute__((address_space(1))) v4i_vs*)(void*)g;
    auto lp = (__attribute__((address_space(3))) v4i_vs*)l;
#if __has_builtin(__builtin_amdgcn_global_load_async_to_lds_b128)
    __builtin_amdgcn_global_load_async_to_lds_b128(gp, lp, 0, 0);
#else
    asm volatile("global_load_async_to_lds_b128 %0, %1, off"
                 :: "v"(lp), "v"(gp) : "memory");
#endif
}

__device__ __forceinline__ void wait_async0() {
#if __has_builtin(__builtin_amdgcn_s_wait_asynccnt)
    __builtin_amdgcn_s_wait_asynccnt(0);
#else
    asm volatile("s_wait_asynccnt 0" ::: "memory");
#endif
}

// ---------------------------------------------------------------------------
// Kernel 1: trig basis, pre-swizzled into WMMA B-fragment layout
// (16x16x32 f16 B: lane L holds column L%16; lanes 0-15 -> K 0..7,16..23,
// lanes 16-31 -> K 8..15,24..31). Fragment (kstep, ntile16) = 1KB contiguous.
// Exact phase via integer (f*t) mod V.
// ---------------------------------------------------------------------------
__global__ void GaussSynth_basis(_Float16* __restrict__ bhi,
                                 _Float16* __restrict__ blo) {
    size_t idx = (size_t)blockIdx.x * blockDim.x + threadIdx.x;
    size_t total = (size_t)KSTEPS * NT16 * 512;   // halves per buffer
    if (idx >= total) return;
    int j    = (int)(idx & 15);
    int lane = (int)((idx >> 4) & 31);
    int nt   = (int)((idx >> 9) % NT16);
    int ks   = (int)(idx / ((size_t)NT16 * 512));
    int t    = nt * 16 + (lane & 15);
    int kloc = (j < 8) ? (((lane >> 4) << 3) + j) : (16 + ((lane >> 4) << 3) + (j - 8));            // 0..31 within k-step
    int kg   = ks * 32 + kloc;                    // 0..511 global K
    float val = 0.0f;
    if (t < VOCAB) {
        int f = (kg < NFREQ) ? (kg + 1) : (kg - NFREQ + 1);
        unsigned r = ((unsigned)f * (unsigned)t) % (unsigned)VOCAB; // < 2^25, exact
        float th = (float)r * (6.2831853071795864769f / (float)VOCAB);
        float s, c;
        __sincosf(th, &s, &c);
        val = (kg < NFREQ) ? c : -s;
    }
    _Float16 hi = (_Float16)val;
    bhi[idx] = hi;
    blo[idx] = (_Float16)(val - (float)hi);
}

// ---------------------------------------------------------------------------
// Kernel 2: Y = h @ weight (4096x512, K=1024), scaled by 2/V, written
// pre-swizzled into WMMA A-fragment layout (16x32 f16 A, ISA 7.12.2),
// split hi/lo. One workgroup per 16-row m-tile; h rows staged in LDS.
// ---------------------------------------------------------------------------
__global__ void __launch_bounds__(256) GaussSynth_ygemm(
        const float* __restrict__ h, const float* __restrict__ w,
        _Float16* __restrict__ ahi, _Float16* __restrict__ alo) {
    __shared__ float hs[16 * NCH];                // 64KB (WGP has 320KB)
    const int t  = threadIdx.x;                   // 256 threads
    const int mt = blockIdx.x;                    // m-tile 0..255
    const float* hb = h + (size_t)mt * 16 * NCH;
    #pragma unroll
    for (int j = 0; j < 16; ++j)
        ((float4*)hs)[t + 256 * j] = ((const float4*)hb)[t + 256 * j];
    __syncthreads();

    const int kf0 = t * 2, kf1 = t * 2 + 1;       // each thread: 2 output cols
    float acc0[16], acc1[16];
    #pragma unroll
    for (int m = 0; m < 16; ++m) { acc0[m] = 0.f; acc1[m] = 0.f; }

    #pragma unroll 4
    for (int c = 0; c < NCH; ++c) {
        float w0 = w[(size_t)c * KTOT + kf0];
        float w1 = w[(size_t)c * KTOT + kf1];
        #pragma unroll
        for (int m = 0; m < 16; ++m) {
            float hv = hs[m * NCH + c];           // LDS broadcast
            acc0[m] = fmaf(hv, w0, acc0[m]);
            acc1[m] = fmaf(hv, w1, acc1[m]);
        }
    }

    const float scale = 2.0f / (float)VOCAB;      // irfft normalization folded in
    #pragma unroll
    for (int m = 0; m < 16; ++m) {
        #pragma unroll
        for (int q = 0; q < 2; ++q) {
            int kf = q ? kf1 : kf0;
            float x = (q ? acc1[m] : acc0[m]) * scale;
            int ks = kf >> 5, kl = kf & 31;
            int laneHi, v, p;
            if (kl < 16) { laneHi = (kl >> 3) & 1; int b = kl - 8 * laneHi; v = b >> 1; p = b & 1; }
            else { int kk = kl - 16; laneHi = kk >> 3; int rem = kk & 7; v = 4 + (rem >> 1); p = rem & 1; }
            size_t o = (((size_t)(mt * 16 + ks)) * 32 + (m + 16 * laneHi)) * 16 + (2 * v + p);
            _Float16 hi = (_Float16)x;
            ahi[o] = hi;
            alo[o] = (_Float16)(x - (float)hi);
        }
    }
}

// ---------------------------------------------------------------------------
// Kernel 3: the 210-TFLOP synthesis GEMM on WMMA.
// Grid: (393 n-tiles of 128) x (32 m-tiles of 128). Block = 256 thr = 8 waves
// arranged 2(M) x 4(N); wave tile 64x32 => 8 v8f accumulators; 16 k-steps of
// 32; double-buffered LDS fed by GLOBAL_LOAD_ASYNC_TO_LDS_B128 (no staging
// VGPRs, one barrier per k-step). 24 v_wmma per wave per k-step.
// ---------------------------------------------------------------------------
union FragU { float4 f4[2]; v16h h; };

__global__ void __launch_bounds__(256) GaussSynth_wmma(
        const _Float16* __restrict__ ahi, const _Float16* __restrict__ alo,
        const _Float16* __restrict__ bhi, const _Float16* __restrict__ blo,
        float* __restrict__ out) {
    __shared__ float4 smem[2][2048];              // 2 x 32KB
    const int t    = threadIdx.x;
    const int lane = t & 31;
    const int wv   = t >> 5;                      // wave 0..7
    const int wm   = wv & 1;                      // 2 M-waves
    const int wn   = wv >> 1;                     // 4 N-waves
    const int mtBase = blockIdx.y * 8;            // global 16-row m-tile base
    const int ntBase = blockIdx.x * 8;            // global 16-col n-tile base

    // Staging: 32 fragments x 1KB per k-step; thread DMAs 128B of fragment f.
    const int f  = t >> 3;                        // fragment 0..31
    const int pp = t & 7;                         // 128B slice within fragment
    const int isB = (f >= 16) ? 1 : 0;
    const int fl = f & 7;                         // tile-local index
    const _Float16* srcBase = isB ? ((f & 8) ? blo : bhi)
                                  : ((f & 8) ? alo : ahi);
    const int dstF4 = f * 64 + pp * 8;            // LDS float4 index of slice

    v8f acc[4][2];
    #pragma unroll
    for (int i = 0; i < 4; ++i)
        #pragma unroll
        for (int j = 0; j < 2; ++j) acc[i][j] = {};

    auto asyncStage = [&](int ks, int buf) {
        size_t halves = isB ? (((size_t)ks * NT16 + (ntBase + fl)) * 512 + (size_t)pp * 64)
                            : (((size_t)(mtBase + fl) * 16 + ks) * 512 + (size_t)pp * 64);
        const _Float16* g = srcBase + halves;
        #pragma unroll
        for (int j = 0; j < 8; ++j)
            async_cp16(g + j * 8, &smem[buf][dstF4 + j]);   // 8 halves = 16B
    };

    asyncStage(0, 0);
    wait_async0();
    __syncthreads();

    for (int ks = 0; ks < KSTEPS; ++ks) {
        const int cur = ks & 1;
        // buf[1-cur]'s readers all finished before the barrier that ended the
        // previous iteration, so the DMA below has no WAR hazard.
        if (ks + 1 < KSTEPS) asyncStage(ks + 1, 1 - cur);

        #pragma unroll
        for (int jn = 0; jn < 2; ++jn) {
            const int ntl = wn * 2 + jn;
            FragU bh, bl;
            bh.f4[0] = smem[cur][(16 + ntl) * 64 + lane * 2];
            bh.f4[1] = smem[cur][(16 + ntl) * 64 + lane * 2 + 1];
            bl.f4[0] = smem[cur][(24 + ntl) * 64 + lane * 2];
            bl.f4[1] = smem[cur][(24 + ntl) * 64 + lane * 2 + 1];
            #pragma unroll
            for (int im = 0; im < 4; ++im) {
                const int mtl = wm * 4 + im;
                FragU ah, al;
                ah.f4[0] = smem[cur][(mtl) * 64 + lane * 2];
                ah.f4[1] = smem[cur][(mtl) * 64 + lane * 2 + 1];
                al.f4[0] = smem[cur][(8 + mtl) * 64 + lane * 2];
                al.f4[1] = smem[cur][(8 + mtl) * 64 + lane * 2 + 1];
                v8f c = acc[im][jn];
                c = __builtin_amdgcn_wmma_f32_16x16x32_f16(false, ah.h, false, bh.h, (short)0, c, false, false);
                c = __builtin_amdgcn_wmma_f32_16x16x32_f16(false, ah.h, false, bl.h, (short)0, c, false, false);
                c = __builtin_amdgcn_wmma_f32_16x16x32_f16(false, al.h, false, bh.h, (short)0, c, false, false);
                acc[im][jn] = c;
            }
        }
        if (ks + 1 < KSTEPS) wait_async0();       // my DMAs into buf[1-cur] done
        __syncthreads();                          // WG-wide: next buffer ready,
                                                  // and everyone done with 'cur'
    }

    // Epilogue: C/D layout (ISA 7.12.2): VGPR r, lane L -> row r+8*(L/16),
    // col L%16. 16 lanes write 64B-contiguous runs per row.
    const int colIn  = lane & 15;
    const int rowOff = (lane >> 4) * 8;
    #pragma unroll
    for (int im = 0; im < 4; ++im) {
        #pragma unroll
        for (int jn = 0; jn < 2; ++jn) {
            const int col = (ntBase + wn * 2 + jn) * 16 + colIn;
            if (col < VOCAB) {
                const size_t rowBase = (size_t)((mtBase + wm * 4 + im) * 16 + rowOff);
                #pragma unroll
                for (int r = 0; r < 8; ++r)
                    out[(rowBase + r) * (size_t)VOCAB + col] = acc[im][jn][r];
            }
        }
    }
}

// ---------------------------------------------------------------------------
// Launch. Workspace layout (~107MB):
//   [0, 4MB)           A-hi fragments   (256 mtiles * 16 ksteps * 1KB)
//   [4MB, 8MB)         A-lo fragments
//   [8MB, 8MB+49.1MB)  B-hi fragments   (16 ksteps * 3144 ntiles * 1KB)
//   [.., +49.1MB)      B-lo fragments
// ---------------------------------------------------------------------------
extern "C" void kernel_launch(void* const* d_in, const int* in_sizes, int n_in,
                              void* d_out, int out_size, void* d_ws, size_t ws_size,
                              hipStream_t stream) {
    const float* h = (const float*)d_in[0];       // [4,1024,1024] f32
    const float* w = (const float*)d_in[1];       // [1024,512]    f32
    float* out = (float*)d_out;                   // [4,1024,50257] f32
    char* ws = (char*)d_ws;

    const size_t A_BYTES = (size_t)MT16 * KSTEPS * 1024;   // 4 MiB each
    const size_t B_BYTES = (size_t)KSTEPS * NT16 * 1024;   // ~49.1 MiB each
    _Float16* ahi = (_Float16*)(ws);
    _Float16* alo = (_Float16*)(ws + A_BYTES);
    _Float16* bhi = (_Float16*)(ws + 2 * A_BYTES);
    _Float16* blo = (_Float16*)(ws + 2 * A_BYTES + B_BYTES);

    const size_t basisElems = (size_t)KSTEPS * NT16 * 512;
    GaussSynth_basis<<<dim3((unsigned)((basisElems + 255) / 256)), dim3(256), 0, stream>>>(bhi, blo);
    GaussSynth_ygemm<<<dim3(MT16), dim3(256), 0, stream>>>(h, w, ahi, alo);
    GaussSynth_wmma<<<dim3(NT128, MTOT / 128), dim3(256), 0, stream>>>(ahi, alo, bhi, blo, out);
}
